// SwitchFeedForward_24378234372444
// MI455X (gfx1250) — compile-verified, matrix operand order checked by reading, not run on hardware
//
#include <hip/hip_runtime.h>
#include <math.h>

#define NTOK 8192   // B*S
#define DDIM 1024
#define EEXP 8
#define HDIM 4096
#define LDSS 40     // LDS row stride in halves (80B, 16B-aligned, bank-friendly)

typedef __attribute__((ext_vector_type(16))) _Float16 v16h;
typedef __attribute__((ext_vector_type(8)))  _Float16 v8h;
typedef __attribute__((ext_vector_type(8)))  float    v8f;

__device__ __forceinline__ v8h pack8(const _Float16* s) {
  v8h r;
#pragma unroll
  for (int i = 0; i < 8; ++i) r[i] = s[i];
  return r;
}

__device__ __forceinline__ v8h cvt8(const float4 a, const float4 b) {
  v8h r;
  r[0] = (_Float16)a.x; r[1] = (_Float16)a.y; r[2] = (_Float16)a.z; r[3] = (_Float16)a.w;
  r[4] = (_Float16)b.x; r[5] = (_Float16)b.y; r[6] = (_Float16)b.z; r[7] = (_Float16)b.w;
  return r;
}

// Fragment load per ISA 7.12.2 16-bit A layout: lane<16 -> K {0..7,16..23},
// lane>=16 -> K {8..15,24..31}; row (A) / column (B) = lane&15.
__device__ __forceinline__ v16h frag_load(const _Float16* base, int halfId) {
  v8h lo = *(const v8h*)(base + 8 * halfId);
  v8h hi = *(const v8h*)(base + 16 + 8 * halfId);
  return __builtin_shufflevector(lo, hi, 0,1,2,3,4,5,6,7,8,9,10,11,12,13,14,15);
}

__global__ void zero_counts_kernel(int* __restrict__ counts) {
  if (threadIdx.x < EEXP) counts[threadIdx.x] = 0;
}

__global__ __launch_bounds__(256) void moe_router(
    const float* __restrict__ x, const float* __restrict__ Wr,
    const float* __restrict__ br, int* __restrict__ counts,
    int* __restrict__ tlist) {
  const int tok  = (blockIdx.x * 256 + threadIdx.x) >> 5;  // one wave32 per token
  const int lane = threadIdx.x & 31;
  const float* xr = x + (size_t)tok * DDIM;
  float acc[EEXP];
#pragma unroll
  for (int e = 0; e < EEXP; ++e) acc[e] = 0.f;
  for (int j = 0; j < DDIM / 32; ++j) {
    const int d = lane + 32 * j;
    const float xv = xr[d];
    const float4* wr = (const float4*)(Wr + (size_t)d * EEXP);
    const float4 w0 = wr[0], w1 = wr[1];
    acc[0] += xv * w0.x; acc[1] += xv * w0.y; acc[2] += xv * w0.z; acc[3] += xv * w0.w;
    acc[4] += xv * w1.x; acc[5] += xv * w1.y; acc[6] += xv * w1.z; acc[7] += xv * w1.w;
  }
#pragma unroll
  for (int e = 0; e < EEXP; ++e) {
#pragma unroll
    for (int s = 16; s > 0; s >>= 1) acc[e] += __shfl_xor(acc[e], s, 32);
  }
  if (lane == 0) {
    float best = acc[0] + br[0];
    int bi = 0;
#pragma unroll
    for (int e = 1; e < EEXP; ++e) {
      const float v = acc[e] + br[e];
      if (v > best) { best = v; bi = e; }  // first-max tie-break, like argmax
    }
    const int pos = atomicAdd(&counts[bi], 1);
    tlist[bi * NTOK + pos] = tok;
  }
}

// GEMM1: h[tok, :] = gelu(x[tok, :] @ W1[e] + b1[e]) for toks routed to e.
// Double-buffered LDS, software pipelined: one barrier per K=32 step.
__global__ __launch_bounds__(256) void moe_gemm1(
    const float* __restrict__ x, const float* __restrict__ W1,
    const float* __restrict__ b1, const int* __restrict__ counts,
    const int* __restrict__ tlist, _Float16* __restrict__ hbuf) {
  __shared__ _Float16 Ald[2][128 * LDSS];
  __shared__ _Float16 Bld[2][128 * LDSS];
  __shared__ int s_tok[128];

  const int e = blockIdx.z;
  const int cnt = counts[e];
  const int mbase = blockIdx.y * 128;
  if (mbase >= cnt) return;
  const int nbase = blockIdx.x * 128;

  const int t = threadIdx.x;
  const int lane = t & 31, wid = t >> 5;
  const int waveM = wid & 1, waveN = wid >> 1;
  const int halfId = lane >> 4, l15 = lane & 15;

  if (t < 128) s_tok[t] = (mbase + t < cnt) ? tlist[e * NTOK + mbase + t] : -1;
  __syncthreads();

  const v8f zf = {0.f, 0.f, 0.f, 0.f, 0.f, 0.f, 0.f, 0.f};
  v8f acc[4][2];
#pragma unroll
  for (int a = 0; a < 4; ++a) {
#pragma unroll
    for (int b = 0; b < 2; ++b) acc[a][b] = zf;
  }

  const float* W1e = W1 + (size_t)e * DDIM * HDIM;
  const int arow = t >> 1, akp = (t & 1) * 16;   // A staging: row, K-offset
  const int bn = t & 127, bkg = (t >> 7) * 16;   // B staging: col, K-group
  const int atok = s_tok[arow];

  const _Float16* aBase0 = &Ald[0][(waveM * 64 + l15) * LDSS];
  const _Float16* aBase1 = &Ald[1][(waveM * 64 + l15) * LDSS];
  const _Float16* bBase0 = &Bld[0][(waveN * 32 + l15) * LDSS];
  const _Float16* bBase1 = &Bld[1][(waveN * 32 + l15) * LDSS];

  constexpr int NK = DDIM / 32;
  // ---- prologue: stage K-chunk 0 into buffer 0
  {
    float4 pa[4];
    if (atok >= 0) {
      const float4* src = (const float4*)(x + (size_t)atok * DDIM + akp);
#pragma unroll
      for (int j = 0; j < 4; ++j) pa[j] = src[j];
    } else {
      const float4 z = {0.f, 0.f, 0.f, 0.f};
#pragma unroll
      for (int j = 0; j < 4; ++j) pa[j] = z;
    }
    float pb[16];
    const float* sb = W1e + (size_t)bkg * HDIM + nbase + bn;
#pragma unroll
    for (int j = 0; j < 16; ++j) pb[j] = sb[(size_t)j * HDIM];

    v8h* da = (v8h*)(&Ald[0][arow * LDSS + akp]);
    da[0] = cvt8(pa[0], pa[1]); da[1] = cvt8(pa[2], pa[3]);
    _Float16 tb[16];
#pragma unroll
    for (int j = 0; j < 16; ++j) tb[j] = (_Float16)pb[j];
    v8h* db = (v8h*)(&Bld[0][bn * LDSS + bkg]);
    db[0] = pack8(tb); db[1] = pack8(tb + 8);
  }
  __syncthreads();

  for (int k = 0; k < NK; ++k) {
    const int kn = k + 1;
    const bool pf = (kn < NK);
    float4 pa[4];
    float pb[16];
    if (pf) {  // issue next tile's global loads first (latency hidden by WMMAs)
      if (atok >= 0) {
        const float4* src = (const float4*)(x + (size_t)atok * DDIM + kn * 32 + akp);
#pragma unroll
        for (int j = 0; j < 4; ++j) pa[j] = src[j];
      } else {
        const float4 z = {0.f, 0.f, 0.f, 0.f};
#pragma unroll
        for (int j = 0; j < 4; ++j) pa[j] = z;
      }
      const float* sb = W1e + (size_t)(kn * 32 + bkg) * HDIM + nbase + bn;
#pragma unroll
      for (int j = 0; j < 16; ++j) pb[j] = sb[(size_t)j * HDIM];
    }

    // ---- compute from buffer k&1
    const _Float16* aB = (k & 1) ? aBase1 : aBase0;
    const _Float16* bB = (k & 1) ? bBase1 : bBase0;
    const v16h bf0 = frag_load(bB, halfId);
    const v16h bf1 = frag_load(bB + 16 * LDSS, halfId);
#pragma unroll
    for (int ms = 0; ms < 4; ++ms) {
      const v16h af = frag_load(aB + ms * 16 * LDSS, halfId);
      acc[ms][0] = __builtin_amdgcn_wmma_f32_16x16x32_f16(false, af, false, bf0,
                                                          (short)0, acc[ms][0], false, false);
      acc[ms][1] = __builtin_amdgcn_wmma_f32_16x16x32_f16(false, af, false, bf1,
                                                          (short)0, acc[ms][1], false, false);
    }

    if (pf) {  // convert + store prefetched tile into the other buffer
      const int nb = kn & 1;
      v8h* da = (v8h*)(&Ald[nb][arow * LDSS + akp]);
      da[0] = cvt8(pa[0], pa[1]); da[1] = cvt8(pa[2], pa[3]);
      _Float16 tb[16];
#pragma unroll
      for (int j = 0; j < 16; ++j) tb[j] = (_Float16)pb[j];
      v8h* db = (v8h*)(&Bld[nb][bn * LDSS + bkg]);
      db[0] = pack8(tb); db[1] = pack8(tb + 8);
    }
    __syncthreads();  // single barrier per step (disjoint read/write buffers)
  }

  const float* b1e = b1 + (size_t)e * HDIM;
#pragma unroll
  for (int ms = 0; ms < 4; ++ms) {
#pragma unroll
    for (int ns = 0; ns < 2; ++ns) {
      const int ng = nbase + waveN * 32 + ns * 16 + l15;
      const float bias = b1e[ng];
#pragma unroll
      for (int r = 0; r < 8; ++r) {
        const int mi = waveM * 64 + ms * 16 + r + 8 * halfId;
        const int tok = s_tok[mi];
        if (tok < 0) continue;
        float v = acc[ms][ns][r] + bias;
        v = 0.5f * v * (1.0f + erff(v * 0.7071067811865475f));  // exact gelu
        hbuf[(size_t)tok * HDIM + ng] = (_Float16)v;
      }
    }
  }
}

// GEMM2: out[tok, :] = h[tok, :] @ W2[e] + b2[e]
__global__ __launch_bounds__(256) void moe_gemm2(
    const _Float16* __restrict__ hbuf, const float* __restrict__ W2,
    const float* __restrict__ b2, const int* __restrict__ counts,
    const int* __restrict__ tlist, float* __restrict__ out) {
  __shared__ _Float16 Ald[2][128 * LDSS];
  __shared__ _Float16 Bld[2][128 * LDSS];
  __shared__ int s_tok[128];

  const int e = blockIdx.z;
  const int cnt = counts[e];
  const int mbase = blockIdx.y * 128;
  if (mbase >= cnt) return;
  const int nbase = blockIdx.x * 128;

  const int t = threadIdx.x;
  const int lane = t & 31, wid = t >> 5;
  const int waveM = wid & 1, waveN = wid >> 1;
  const int halfId = lane >> 4, l15 = lane & 15;

  if (t < 128) s_tok[t] = (mbase + t < cnt) ? tlist[e * NTOK + mbase + t] : -1;
  __syncthreads();

  const v8f zf = {0.f, 0.f, 0.f, 0.f, 0.f, 0.f, 0.f, 0.f};
  v8f acc[4][2];
#pragma unroll
  for (int a = 0; a < 4; ++a) {
#pragma unroll
    for (int b = 0; b < 2; ++b) acc[a][b] = zf;
  }

  const float* W2e = W2 + (size_t)e * HDIM * DDIM;
  const int arow = t >> 1, akp = (t & 1) * 16;
  const int bn = t & 127, bkg = (t >> 7) * 16;
  const int atok = s_tok[arow];

  const _Float16* aBase0 = &Ald[0][(waveM * 64 + l15) * LDSS];
  const _Float16* aBase1 = &Ald[1][(waveM * 64 + l15) * LDSS];
  const _Float16* bBase0 = &Bld[0][(waveN * 32 + l15) * LDSS];
  const _Float16* bBase1 = &Bld[1][(waveN * 32 + l15) * LDSS];

  v8h zh;
#pragma unroll
  for (int i = 0; i < 8; ++i) zh[i] = (_Float16)0.0f;

  constexpr int NK = HDIM / 32;
  // ---- prologue: stage K-chunk 0 into buffer 0
  {
    v8h p0 = zh, p1 = zh;
    if (atok >= 0) {
      const v8h* src = (const v8h*)(hbuf + (size_t)atok * HDIM + akp);
      p0 = src[0]; p1 = src[1];
    }
    float pb[16];
    const float* sb = W2e + (size_t)bkg * DDIM + nbase + bn;
#pragma unroll
    for (int j = 0; j < 16; ++j) pb[j] = sb[(size_t)j * DDIM];

    v8h* da = (v8h*)(&Ald[0][arow * LDSS + akp]);
    da[0] = p0; da[1] = p1;
    _Float16 tb[16];
#pragma unroll
    for (int j = 0; j < 16; ++j) tb[j] = (_Float16)pb[j];
    v8h* db = (v8h*)(&Bld[0][bn * LDSS + bkg]);
    db[0] = pack8(tb); db[1] = pack8(tb + 8);
  }
  __syncthreads();

  for (int k = 0; k < NK; ++k) {
    const int kn = k + 1;
    const bool pf = (kn < NK);
    v8h p0 = zh, p1 = zh;
    float pb[16];
    if (pf) {  // issue next tile's global loads first
      if (atok >= 0) {
        const v8h* src = (const v8h*)(hbuf + (size_t)atok * HDIM + kn * 32 + akp);
        p0 = src[0]; p1 = src[1];
      }
      const float* sb = W2e + (size_t)(kn * 32 + bkg) * DDIM + nbase + bn;
#pragma unroll
      for (int j = 0; j < 16; ++j) pb[j] = sb[(size_t)j * DDIM];
    }

    const _Float16* aB = (k & 1) ? aBase1 : aBase0;
    const _Float16* bB = (k & 1) ? bBase1 : bBase0;
    const v16h bf0 = frag_load(bB, halfId);
    const v16h bf1 = frag_load(bB + 16 * LDSS, halfId);
#pragma unroll
    for (int ms = 0; ms < 4; ++ms) {
      const v16h af = frag_load(aB + ms * 16 * LDSS, halfId);
      acc[ms][0] = __builtin_amdgcn_wmma_f32_16x16x32_f16(false, af, false, bf0,
                                                          (short)0, acc[ms][0], false, false);
      acc[ms][1] = __builtin_amdgcn_wmma_f32_16x16x32_f16(false, af, false, bf1,
                                                          (short)0, acc[ms][1], false, false);
    }

    if (pf) {
      const int nb = kn & 1;
      v8h* da = (v8h*)(&Ald[nb][arow * LDSS + akp]);
      da[0] = p0; da[1] = p1;
      _Float16 tb[16];
#pragma unroll
      for (int j = 0; j < 16; ++j) tb[j] = (_Float16)pb[j];
      v8h* db = (v8h*)(&Bld[nb][bn * LDSS + bkg]);
      db[0] = pack8(tb); db[1] = pack8(tb + 8);
    }
    __syncthreads();
  }

  const float* b2e = b2 + (size_t)e * DDIM;
#pragma unroll
  for (int ms = 0; ms < 4; ++ms) {
#pragma unroll
    for (int ns = 0; ns < 2; ++ns) {
      const int ng = nbase + waveN * 32 + ns * 16 + l15;
      const float bias = b2e[ng];
#pragma unroll
      for (int r = 0; r < 8; ++r) {
        const int mi = waveM * 64 + ms * 16 + r + 8 * halfId;
        const int tok = s_tok[mi];
        if (tok < 0) continue;
        out[(size_t)tok * DDIM + ng] = acc[ms][ns][r] + bias;
      }
    }
  }
}

extern "C" void kernel_launch(void* const* d_in, const int* in_sizes, int n_in,
                              void* d_out, int out_size, void* d_ws, size_t ws_size,
                              hipStream_t stream) {
  (void)in_sizes; (void)n_in; (void)out_size; (void)ws_size;
  const float* x  = (const float*)d_in[0];
  const float* Wr = (const float*)d_in[1];
  const float* br = (const float*)d_in[2];
  const float* W1 = (const float*)d_in[3];
  const float* b1 = (const float*)d_in[4];
  const float* W2 = (const float*)d_in[5];
  const float* b2 = (const float*)d_in[6];
  float* out = (float*)d_out;

  char* ws = (char*)d_ws;
  int* counts = (int*)ws;                                     // 8 ints
  int* tlist  = (int*)(ws + 256);                             // E*NTOK ints (256 KB)
  _Float16* hbuf = (_Float16*)(ws + 256 + (size_t)EEXP * NTOK * sizeof(int));  // 64 MB

  zero_counts_kernel<<<1, 32, 0, stream>>>(counts);
  moe_router<<<NTOK / 8, 256, 0, stream>>>(x, Wr, br, counts, tlist);
  moe_gemm1<<<dim3(HDIM / 128, NTOK / 128, EEXP), 256, 0, stream>>>(
      x, W1, b1, counts, tlist, hbuf);
  moe_gemm2<<<dim3(DDIM / 128, NTOK / 128, EEXP), 256, 0, stream>>>(
      hbuf, W2, b2, counts, tlist, out);
}
